// MappingNet_9431748182734
// MI455X (gfx1250) — compile-verified, matrix operand order checked by reading, not run on hardware
//
#include <hip/hip_runtime.h>
#include <hip/hip_bf16.h>
#include <math.h>

// ---------- types ----------
typedef __bf16 bf16_t;
typedef bf16_t bf16x8 __attribute__((ext_vector_type(8)));
typedef bf16_t v16bf  __attribute__((ext_vector_type(16)));
typedef float  v8f    __attribute__((ext_vector_type(8)));
typedef float  f32x4  __attribute__((ext_vector_type(4)));
typedef unsigned int u32x4 __attribute__((ext_vector_type(4)));
typedef int    i32x4  __attribute__((ext_vector_type(4)));
typedef int    i32x8  __attribute__((ext_vector_type(8)));

#define NBATCH   4096
#define IN_DIM   512
#define HID_DIM  1024
#define NGENRE   10

#define BM 128
#define BN 128
#define BK 32
#define LDA 48   // bf16 elems; 96B row stride -> 16B aligned, bank-rotating
#define LDB 48

__device__ __forceinline__ float gelu_exact(float x) {
    return 0.5f * x * (1.0f + erff(x * 0.70710678118654752f));
}

// Issue a TDM 2D tile load: [rows x 32] bf16 tile, row stride K elems, into LDS
// at ldsAddr with padded layout (16 DWORDs data + 8 DWORDs pad -> LDA=48 elems).
__device__ __forceinline__ void tdm_load_tile(const bf16_t* gsrc, unsigned int ldsAddr,
                                              int K, int validRows) {
    unsigned long long ga = (unsigned long long)(uintptr_t)gsrc;
    u32x4 g0;
    g0[0] = 1u;                                   // count=1, user descriptor
    g0[1] = ldsAddr;                              // lds_addr [63:32]
    g0[2] = (unsigned int)(ga & 0xFFFFFFFFull);   // global_addr[31:0]
    g0[3] = (unsigned int)((ga >> 32) & 0x1FFFFFFull) | (2u << 30); // addr[56:32], type=2
    i32x8 g1;
    g1[0] = (int)((1u << 16)     // data_size = 1 -> 2 bytes
                | (1u << 20)     // pad_enable
                | (3u << 22)     // pad_interval: 16 DWORDs (64B row)
                | (7u << 25));   // pad_amount: 8 DWORDs (32B -> stride 96B = LDA)
    g1[1] = (int)(((unsigned)K & 0xFFFFu) << 16);                       // tensor_dim0 lo
    g1[2] = (int)((((unsigned)K >> 16) & 0xFFFFu)
                | (((unsigned)validRows & 0xFFFFu) << 16));             // dim0 hi | dim1 lo
    g1[3] = (int)(((((unsigned)validRows >> 16)) & 0xFFFFu)
                | ((unsigned)BK << 16));                                // dim1 hi | tile_dim0
    g1[4] = BM;          // tile_dim1 = 128 rows (OOB rows read as zero)
    g1[5] = K;           // tensor_dim0_stride lo (elems)
    g1[6] = 0;           // stride hi | tensor_dim1_stride lo (unused, 2D tile)
    g1[7] = 0;
    i32x4 gz;
    gz[0] = 0; gz[1] = 0; gz[2] = 0; gz[3] = 0;
#if defined(__clang_major__) && (__clang_major__ >= 23)
    i32x8 gz8;
    gz8[0] = 0; gz8[1] = 0; gz8[2] = 0; gz8[3] = 0;
    gz8[4] = 0; gz8[5] = 0; gz8[6] = 0; gz8[7] = 0;
    __builtin_amdgcn_tensor_load_to_lds(g0, g1, gz, gz, gz8, 0);
#else
    __builtin_amdgcn_tensor_load_to_lds(g0, g1, gz, gz, 0);
#endif
}

// ---------- prep: noise_genre = mu[g] + (|sigma[g]|+eps)*z  (emit bf16) ----------
__global__ void k_noise(const int* __restrict__ genre, const float* __restrict__ z,
                        const float* __restrict__ mu, const float* __restrict__ sg,
                        bf16_t* __restrict__ X) {
    int idx = blockIdx.x * blockDim.x + threadIdx.x;   // over B*IN_DIM
    int b = idx >> 9;            // /512
    int d = idx & 511;
    int g = genre[b];
    float m = mu[g * IN_DIM + d];
    float s = fabsf(sg[g * IN_DIM + d]) + 1e-8f;
    X[idx] = (bf16_t)(m + s * z[idx]);
}

// ---------- counting sort by genre ----------
__global__ void k_hist(const int* __restrict__ genre, int* __restrict__ cnt) {
    int b = blockIdx.x * blockDim.x + threadIdx.x;
    if (b < NBATCH) atomicAdd(&cnt[genre[b]], 1);
}

__global__ void k_offsets(const int* __restrict__ cnt, int* __restrict__ off) {
    if (threadIdx.x == 0) {
        int s = 0;
        for (int e = 0; e < NGENRE; ++e) { off[e] = s; s += cnt[e]; }
    }
}

__global__ void k_scatter(const int* __restrict__ genre, const int* __restrict__ off,
                          int* __restrict__ cur, int* __restrict__ perm) {
    int b = blockIdx.x * blockDim.x + threadIdx.x;
    if (b < NBATCH) {
        int g = genre[b];
        int p = atomicAdd(&cur[g], 1);
        perm[off[g] + p] = b;
    }
}

// ---------- generic WMMA bf16 GEMM ----------
// MODE 0: C = gelu(A @ W + b)          A rows direct (TDM-staged), out bf16
// MODE 1: per-genre, A rows gathered via perm (vector path), out bf16 compact, gelu
// MODE 2: per-genre, A rows compact (TDM-staged, OOB rows=0), out fp32 scatter, no act
template <int MODE>
__global__ __launch_bounds__(256)
void gemm_wmma(const bf16_t* __restrict__ A, const float* __restrict__ W,
               const float* __restrict__ bias,
               bf16_t* __restrict__ outB, float* __restrict__ outF,
               int M, int K, int N,
               const int* __restrict__ perm, const int* __restrict__ grpOff,
               const int* __restrict__ grpCnt) {
    __shared__ __align__(16) bf16_t As[BM * LDA];
    __shared__ __align__(16) bf16_t Bs[BN * LDB];   // stored K-transposed: Bs[n][k]

    int segBase = 0, cnt = M;
    int m0 = blockIdx.y * BM;
    int n0 = blockIdx.x * BN;
    if (MODE != 0) {
        int g = blockIdx.z;
        segBase = grpOff[g];
        cnt = grpCnt[g];
        if (m0 >= cnt) return;
        W    += (size_t)g * K * N;
        bias += (size_t)g * N;
    }

    int tid  = threadIdx.x;
    int wave = tid >> 5;
    int lane = tid & 31;
    int half = lane >> 4;     // 0: K 0-7/16-23, 1: K 8-15/24-31 (ISA bf16 layout)
    int r    = lane & 15;
    int wm   = (wave >> 1) * 32;   // wave row offset in tile (4 rows of waves)
    int wn   = (wave & 1) * 64;    // wave col offset (2 cols of waves)

    v8f acc[2][4];
#pragma unroll
    for (int i = 0; i < 2; ++i)
#pragma unroll
        for (int j = 0; j < 4; ++j)
#pragma unroll
            for (int v = 0; v < 8; ++v) acc[i][j][v] = 0.0f;

    const int ksteps = K / BK;
    for (int ks = 0; ks < ksteps; ++ks) {
        const int k0 = ks * BK;

        // ---- stage A tile: 128 x 32 bf16 ----
        if (MODE != 1) {
            // Contiguous rows: DMA via Tensor Data Mover (wave 0 issues; DMA
            // overlaps with the B-tile staging done by all 8 waves below).
            if (wave == 0) {
                int rowBase    = (MODE == 0) ? m0 : (segBase + m0);
                int validRows  = (MODE == 0) ? BM : (cnt - m0);  // OOB rows -> 0
                tdm_load_tile(A + (size_t)rowBase * K + k0,
                              (unsigned int)(uintptr_t)(&As[0]), K, validRows);
            }
        } else {
            // Gathered rows via perm: vector path (512 chunks of 8 bf16)
            for (int c = tid; c < (BM * BK / 8); c += 256) {
                int row = c >> 2;
                int kc  = (c & 3) * 8;
                int i = m0 + row;
                bool valid = (i < cnt);
                bf16x8 v;
                if (valid) {
                    size_t arow = (size_t)perm[segBase + i];
                    v = *(const bf16x8*)(A + arow * (size_t)K + k0 + kc);
                } else {
#pragma unroll
                    for (int t = 0; t < 8; ++t) v[t] = (bf16_t)0.0f;
                }
                *(bf16x8*)(&As[row * LDA + kc]) = v;
            }
        }

        // ---- stage B tile: 32 x 128 fp32 -> bf16, K-transposed in LDS ----
        for (int c = tid; c < (BK * BN / 4); c += 256) {
            int kr = c >> 5;            // 0..31
            int nc = (c & 31) * 4;      // 0..124
            const float* wp = W + (size_t)(k0 + kr) * N + n0 + nc;
            f32x4 w = *(const f32x4*)wp;
            if (ks + 1 < ksteps)
                __builtin_prefetch((const void*)(wp + (size_t)BK * N), 0, 3);
#pragma unroll
            for (int jj = 0; jj < 4; ++jj)
                Bs[(nc + jj) * LDB + kr] = (bf16_t)w[jj];
        }

        if (MODE != 1) {
            if (wave == 0) __builtin_amdgcn_s_wait_tensorcnt(0);
        }
        __syncthreads();

        // ---- fragments + WMMA ----
        v16bf afr[2], bfr[4];
#pragma unroll
        for (int i = 0; i < 2; ++i) {
            int row = wm + i * 16 + r;
            bf16x8 lo = *(const bf16x8*)(&As[row * LDA + half * 8]);
            bf16x8 hi = *(const bf16x8*)(&As[row * LDA + 16 + half * 8]);
#pragma unroll
            for (int t = 0; t < 8; ++t) { afr[i][t] = lo[t]; afr[i][t + 8] = hi[t]; }
        }
#pragma unroll
        for (int j = 0; j < 4; ++j) {
            int col = wn + j * 16 + r;
            bf16x8 lo = *(const bf16x8*)(&Bs[col * LDB + half * 8]);
            bf16x8 hi = *(const bf16x8*)(&Bs[col * LDB + 16 + half * 8]);
#pragma unroll
            for (int t = 0; t < 8; ++t) { bfr[j][t] = lo[t]; bfr[j][t + 8] = hi[t]; }
        }
#pragma unroll
        for (int i = 0; i < 2; ++i)
#pragma unroll
            for (int j = 0; j < 4; ++j)
                acc[i][j] = __builtin_amdgcn_wmma_f32_16x16x32_bf16(
                    false, afr[i], false, bfr[j], (short)0, acc[i][j], false, false);
        __syncthreads();
    }

    // ---- epilogue: bias + (gelu) + store (C layout: vgpr v -> row half*8+v) ----
#pragma unroll
    for (int i = 0; i < 2; ++i) {
#pragma unroll
        for (int v = 0; v < 8; ++v) {
            int lrow = wm + i * 16 + half * 8 + v;       // local row in 128-tile
            bool valid = true;
            size_t orow = 0;
            if (MODE == 0) {
                orow = (size_t)(m0 + lrow);
            } else if (MODE == 1) {
                valid = (m0 + lrow) < cnt;
                orow = (size_t)(segBase + m0 + lrow);    // compact store
            } else {
                valid = (m0 + lrow) < cnt;
                orow = valid ? (size_t)perm[segBase + m0 + lrow] : 0;
            }
            if (!valid) continue;
#pragma unroll
            for (int j = 0; j < 4; ++j) {
                int col = n0 + wn + j * 16 + r;
                float x = acc[i][j][v] + bias[col];
                if (MODE == 2) {
                    outF[orow * (size_t)N + col] = x;
                } else {
                    outB[orow * (size_t)N + col] = (bf16_t)gelu_exact(x);
                }
            }
        }
    }
}

// ---------- launch ----------
extern "C" void kernel_launch(void* const* d_in, const int* in_sizes, int n_in,
                              void* d_out, int out_size, void* d_ws, size_t ws_size,
                              hipStream_t stream) {
    (void)in_sizes; (void)n_in; (void)out_size; (void)ws_size;
    const int*   genre = (const int*)  d_in[1];
    const float* z     = (const float*)d_in[2];
    const float* mu    = (const float*)d_in[3];
    const float* sg    = (const float*)d_in[4];
    const float* Ws1   = (const float*)d_in[5];
    const float* bs1   = (const float*)d_in[6];
    const float* Ws2   = (const float*)d_in[7];
    const float* bs2   = (const float*)d_in[8];
    const float* We1   = (const float*)d_in[9];
    const float* be1   = (const float*)d_in[10];
    const float* We2   = (const float*)d_in[11];
    const float* be2   = (const float*)d_in[12];
    float* out = (float*)d_out;

    char* ws = (char*)d_ws;
    bf16_t* X  = (bf16_t*)ws;                              //  4 MB [4096x512]
    bf16_t* S1 = (bf16_t*)(ws + ((size_t)4 << 20));        //  8 MB [4096x1024]
    bf16_t* S2 = (bf16_t*)(ws + ((size_t)12 << 20));       //  8 MB [4096x1024]
    bf16_t* H  = S1;                                       // reuse S1 after gemm2
    int* cnt  = (int*)(ws + ((size_t)20 << 20));
    int* off  = cnt + 16;
    int* cur  = cnt + 32;
    int* perm = cnt + 48;

    hipMemsetAsync(cnt, 0, 48 * sizeof(int), stream);

    k_noise  <<<(NBATCH * IN_DIM) / 256, 256, 0, stream>>>(genre, z, mu, sg, X);
    k_hist   <<<NBATCH / 256, 256, 0, stream>>>(genre, cnt);
    k_offsets<<<1, 32, 0, stream>>>(cnt, off);
    k_scatter<<<NBATCH / 256, 256, 0, stream>>>(genre, off, cur, perm);

    dim3 blk(256);
    dim3 gShared(HID_DIM / BN, NBATCH / BM);               // 8 x 32
    gemm_wmma<0><<<gShared, blk, 0, stream>>>(X,  Ws1, bs1, S1, nullptr,
                                              NBATCH, IN_DIM, HID_DIM,
                                              nullptr, nullptr, nullptr);
    gemm_wmma<0><<<gShared, blk, 0, stream>>>(S1, Ws2, bs2, S2, nullptr,
                                              NBATCH, HID_DIM, HID_DIM,
                                              nullptr, nullptr, nullptr);

    dim3 gExp(HID_DIM / BN, NBATCH / BM, NGENRE);          // 8 x 32 x 10 (early-exit)
    gemm_wmma<1><<<gExp, blk, 0, stream>>>(S2, We1, be1, H, nullptr,
                                           NBATCH, HID_DIM, HID_DIM,
                                           perm, off, cnt);
    gemm_wmma<2><<<gExp, blk, 0, stream>>>(H,  We2, be2, nullptr, out,
                                           NBATCH, HID_DIM, HID_DIM,
                                           perm, off, cnt);
}
